// DFAttention_45578192945777
// MI455X (gfx1250) — matrix operand order repaired and checked by reading, not compiled, not run on hardware
//
#include <hip/hip_runtime.h>

#define B_  256
#define N_  256
#define C_  192
#define H_  6
#define HD_ 32
#define AG_ 49
#define AP_ 64
#define SCALE_ 0.17677669529663689f   // 32^-0.5

typedef unsigned short u16b;          // raw bf16 bits
typedef __attribute__((ext_vector_type(16))) __bf16 v16bf;
typedef __attribute__((ext_vector_type(8)))  float  v8f;
typedef __attribute__((ext_vector_type(4)))  int    i32x4;

#if __has_builtin(__builtin_amdgcn_global_load_async_to_lds_b128) && \
    __has_builtin(__builtin_amdgcn_s_wait_asynccnt)
#define HAVE_ASYNC_LDS 1
#define AS1G __attribute__((address_space(1)))
#define AS3L __attribute__((address_space(3)))
#endif

// ---------------- helpers ----------------

static __device__ __forceinline__ u16b f2bf(float f) {
  unsigned u = __float_as_uint(f);
  u += 0x7FFFu + ((u >> 16) & 1u);    // round-to-nearest-even
  return (u16b)(u >> 16);
}

// 16 contiguous bf16 from p (B fragment: K contiguous per lane)
static __device__ __forceinline__ v16bf frag_contig(const u16b* p) {
  union { uint4 u[2]; v16bf v; } f;
  f.u[0] = *reinterpret_cast<const uint4*>(p);
  f.u[1] = *reinterpret_cast<const uint4*>(p + 8);
  return f.v;
}

// A fragment for a 16-wide row slice of a 32-wide K window starting at row32.
// lanes 0-15: K = 0..15 ; lanes 16-31: K = 16..31
static __device__ __forceinline__ v16bf frag_A(const u16b* row32, int lane) {
  int kb = (lane & 16) ? 16 : 0;
  union { uint4 u[2]; v16bf v; } f;
  f.u[0] = *reinterpret_cast<const uint4*>(row32 + kb);
  f.u[1] = *reinterpret_cast<const uint4*>(row32 + kb + 8);
  return f.v;
}

static __device__ __forceinline__ v8f wmma_bf16(v16bf a, v16bf b, v8f c) {
  return __builtin_amdgcn_wmma_f32_16x16x32_bf16(false, a, false, b, (short)0, c,
                                                 false, false);
}

// bilinear sample of a 7x7 image at output pixel (i,j) of a 16x16 grid,
// align_corners=False semantics
static __device__ __forceinline__ float bilin7(const float* img, int i, int j) {
  float si = (i + 0.5f) * (7.0f / 16.0f) - 0.5f; si = fmaxf(si, 0.0f);
  float sj = (j + 0.5f) * (7.0f / 16.0f) - 0.5f; sj = fmaxf(sj, 0.0f);
  int i0 = min((int)floorf(si), 6); int i1 = min(i0 + 1, 6); float ti = si - (float)i0;
  int j0 = min((int)floorf(sj), 6); int j1 = min(j0 + 1, 6); float tj = sj - (float)j0;
  float a = img[i0 * 7 + j0] * (1.0f - tj) + img[i0 * 7 + j1] * tj;
  float b = img[i1 * 7 + j0] * (1.0f - tj) + img[i1 * 7 + j1] * tj;
  return a * (1.0f - ti) + b * ti;
}

// ---------------- elementwise prep: bf16 GEMM operands ----------------

__global__ __launch_bounds__(256) void prep_k(
    const float* __restrict__ xF1, const float* __restrict__ xC2,
    const float* __restrict__ xC1,
    u16b* __restrict__ Xq, u16b* __restrict__ Xk, u16b* __restrict__ Xv,
    u16b* __restrict__ Xc1, u16b* __restrict__ Xc2) {
  int i = blockIdx.x * 256 + threadIdx.x;
  if (i >= B_ * N_ * C_) return;
  float f1 = xF1[i], c2 = xC2[i], c1 = xC1[i];
  Xq[i]  = f2bf(f1);
  Xk[i]  = f2bf(f1 + c2 - c1);
  Xv[i]  = f2bf(c2 - c1);
  Xc1[i] = f2bf(c1);
  Xc2[i] = f2bf(c2);
}

// ---------------- weight pack: W (NxK row-major) -> per-lane B fragments ----
// layout: [kk(6)][nt(12)][lane(32)][i(16)]  with  n = nt*16 + lane%16,
// k = kk*32 + (lane<16 ? i : 16+i)

__global__ __launch_bounds__(256) void pack_w_k(const float* __restrict__ W,
                                                u16b* __restrict__ Wpk) {
  int idx = blockIdx.x * 256 + threadIdx.x;
  if (idx >= 6 * 12 * 32 * 16) return;
  int i = idx & 15;
  int lane = (idx >> 4) & 31;
  int nt = (idx >> 9) % 12;
  int kk = idx / (16 * 32 * 12);
  int n = nt * 16 + (lane & 15);
  int k = kk * 32 + ((lane & 16) ? 16 + i : i);
  Wpk[idx] = f2bf(W[n * C_ + k]);
}

// ---------------- bf16 GEMM:  Y[M x 192] = X[M x 192] @ Wpk + bias ---------
// one wave per 16-row strip: A fragment loaded once per k-step, reused for
// all 12 n-tiles (12 f32 accumulators live in VGPRs).

template <int OUT_BF16>
__global__ __launch_bounds__(256) void gemm2_k(
    const u16b* __restrict__ X, const u16b* __restrict__ Wpk,
    const float* __restrict__ bias, void* __restrict__ Y, int M) {
  int wave = threadIdx.x >> 5, lane = threadIdx.x & 31;
  int strip = blockIdx.x * 8 + wave;
  if (strip >= (M >> 4)) return;
  int row = (strip << 4) + (lane & 15);

  // prefetch the strip this block will touch 16 strips ahead (next wavefront
  // of blocks) -> global_prefetch_b8, keeps HBM stream ahead of WMMA
  if (row + 256 < M) __builtin_prefetch(X + (size_t)(row + 256) * C_, 0, 3);

  v8f acc[12];
#pragma unroll
  for (int nt = 0; nt < 12; ++nt) acc[nt] = {};

  const u16b* wl = Wpk + (lane << 4);
#pragma unroll
  for (int kk = 0; kk < 6; ++kk) {
    v16bf a = frag_A(X + (size_t)row * C_ + kk * 32, lane);
    const u16b* wk = wl + kk * 6144;  // kk*12*32*16
#pragma unroll
    for (int nt = 0; nt < 12; ++nt) {
      v16bf b = frag_contig(wk + nt * 512);  // nt*32*16
      acc[nt] = wmma_bf16(a, b, acc[nt]);
    }
  }

  int mbase = (strip << 4) + ((lane & 16) ? 8 : 0);
#pragma unroll
  for (int nt = 0; nt < 12; ++nt) {
    int n = (nt << 4) + (lane & 15);
    float bv = bias[n];
#pragma unroll
    for (int r = 0; r < 8; ++r) {
      float val = acc[nt][r] + bv;
      size_t idx = (size_t)(mbase + r) * C_ + n;
      if constexpr (OUT_BF16) ((u16b*)Y)[idx] = f2bf(val);
      else                    ((float*)Y)[idx] = val;
    }
  }
}

// ---------------- adaptive avg pool 16x16 -> 7x7, pad agents to 64 ---------

__global__ __launch_bounds__(256) void pool_k(const float* __restrict__ lin,
                                              float* __restrict__ out) {
  int t = blockIdx.x * 256 + threadIdx.x;
  if (t >= B_ * AP_ * C_) return;
  int c = t % C_, a = (t / C_) % AP_, b = t / (C_ * AP_);
  if (a >= AG_) { out[t] = 0.0f; return; }
  int p = a / 7, q = a % 7;
  int hs = (p * 16) / 7,  he = ((p + 1) * 16 + 6) / 7;
  int ws = (q * 16) / 7,  we = ((q + 1) * 16 + 6) / 7;
  float s = 0.0f;
  for (int hh = hs; hh < he; ++hh)
    for (int ww = ws; ww < we; ++ww)
      s += lin[(b * N_ + hh * 16 + ww) * C_ + c];
  out[t] = s / (float)((he - hs) * (we - ws));
}

// ---------------- positional biases ----------------------------------------

__global__ __launch_bounds__(256) void bias_pb_k(
    const float* __restrict__ an, const float* __restrict__ ah,
    const float* __restrict__ aw, float* __restrict__ pb) {
  int t = blockIdx.x * 256 + threadIdx.x;
  if (t >= H_ * AG_ * N_) return;
  int n = t % N_, a = (t / N_) % AG_, h = t / (N_ * AG_);
  int hh = n >> 4, ww = n & 15;
  float v = bilin7(an + (h * AG_ + a) * 49, hh, ww);
  pb[t] = v + ah[(h * AG_ + a) * 16 + hh] + aw[(h * AG_ + a) * 16 + ww];
}

__global__ __launch_bounds__(256) void bias_ab_k(
    const float* __restrict__ na, const float* __restrict__ ha,
    const float* __restrict__ wa, float* __restrict__ ab) {
  int t = blockIdx.x * 256 + threadIdx.x;
  if (t >= H_ * N_ * AP_) return;
  int a = t % AP_, n = (t / AP_) % N_, h = t / (AP_ * N_);
  if (a >= AG_) { ab[t] = -1e30f; return; }  // mask pad agents
  int hh = n >> 4, ww = n & 15;
  float v = bilin7(na + (h * AG_ + a) * 49, hh, ww);
  ab[t] = v + ha[(h * 16 + hh) * AG_ + a] + wa[(h * 16 + ww) * AG_ + a];
}

// ---------------- stage 1: agent <- k/v attention --------------------------
// one workgroup (8 waves) per (b,h).  agent_v[B][H][64][32] (f32)

__global__ __launch_bounds__(256) void stage1_k(
    const float* __restrict__ a2p, const u16b* __restrict__ kbuf,
    const u16b* __restrict__ vbuf, const float* __restrict__ pb,
    float* __restrict__ agent_v) {
  __shared__ u16b  Klds[N_ * HD_];   // [n][d]  16KB
  __shared__ u16b  VT[HD_ * N_];     // [d][n]  16KB
  __shared__ u16b  A2[AP_ * HD_];    // [a][d]   4KB
  __shared__ float S[16 * N_];       //         16KB
  __shared__ u16b  Sb[16 * N_];      //          8KB
  __shared__ float red[512];         //          2KB
  int b = blockIdx.x / H_, h = blockIdx.x % H_;
  int tid = threadIdx.x, wave = tid >> 5, lane = tid & 31;
  int koff = (lane & 16) ? 16 : 0;

#ifdef HAVE_ASYNC_LDS
  // K tile is a pure bulk copy (bf16 -> bf16, same row layout): use the
  // gfx1250 async global->LDS path, 16B per lane-chunk, tracked by ASYNCcnt.
  for (int ch = tid; ch < (N_ * HD_) / 8; ch += 256) {
    int n = ch >> 2, d0 = (ch & 3) * 8;
    const u16b* src = kbuf + (size_t)(b * N_ + n) * C_ + h * HD_ + d0;
    __builtin_amdgcn_global_load_async_to_lds_b128(
        (AS1G i32x4*)(void*)const_cast<u16b*>(src),
        (AS3L i32x4*)(void*)(Klds + n * HD_ + d0), 0, 0);
  }
  for (int i = tid; i < N_ * HD_; i += 256) {
    int n = i >> 5, d = i & 31;
    VT[d * N_ + n] = vbuf[(b * N_ + n) * C_ + h * HD_ + d];
  }
#else
  for (int i = tid; i < N_ * HD_; i += 256) {
    int n = i >> 5, d = i & 31;
    Klds[i] = kbuf[(b * N_ + n) * C_ + h * HD_ + d];
    VT[d * N_ + n] = vbuf[(b * N_ + n) * C_ + h * HD_ + d];
  }
#endif
  for (int i = tid; i < AP_ * HD_; i += 256)
    A2[i] = f2bf(a2p[(b * AP_ + (i >> 5)) * C_ + h * HD_ + (i & 31)]);
#ifdef HAVE_ASYNC_LDS
  __builtin_amdgcn_s_wait_asynccnt(0);
#endif
  __syncthreads();

  for (int mt = 0; mt < 4; ++mt) {
    // S = SCALE * A2[16 rows] @ K^T + pb   (16 x 256)
    v16bf a = frag_A(A2 + ((mt << 4) + (lane & 15)) * HD_, lane);
    for (int tt = 0; tt < 2; ++tt) {
      int nt = wave * 2 + tt;
      v16bf bfr = frag_contig(Klds + ((nt << 4) + (lane & 15)) * HD_ + koff);
      v8f c = {};
      c = wmma_bf16(a, bfr, c);
      int n = (nt << 4) + (lane & 15);
      int abase = (lane & 16) ? 8 : 0;
#pragma unroll
      for (int r = 0; r < 8; ++r) {
        int al = abase + r, ag = (mt << 4) + al;
        float val = SCALE_ * c[r];
        if (ag < AG_) val += pb[(h * AG_ + ag) * N_ + n];
        S[al * N_ + n] = val;
      }
    }
    __syncthreads();
    // softmax over n (16 rows x 256)
    {
      int rr = tid >> 4, sub = tid & 15;
      float mx = -1e30f;
      for (int j = sub; j < N_; j += 16) mx = fmaxf(mx, S[rr * N_ + j]);
      red[tid] = mx;
      __syncthreads();
      if (sub == 0) {
        float m2 = red[tid];
        for (int i = 1; i < 16; ++i) m2 = fmaxf(m2, red[tid + i]);
        red[tid] = m2;
      }
      __syncthreads();
      float rowmax = red[rr << 4];
      float sm = 0.0f;
      for (int j = sub; j < N_; j += 16) {
        float e = __expf(S[rr * N_ + j] - rowmax);
        S[rr * N_ + j] = e; sm += e;
      }
      red[256 + tid] = sm;
      __syncthreads();
      if (sub == 0) {
        float s2 = 0.0f;
        for (int i = 0; i < 16; ++i) s2 += red[256 + tid + i];
        red[256 + tid] = s2;
      }
      __syncthreads();
      float inv = 1.0f / red[256 + (rr << 4)];
      for (int j = sub; j < N_; j += 16)
        Sb[rr * N_ + j] = f2bf(S[rr * N_ + j] * inv);
    }
    __syncthreads();
    // agent_v[16 rows] = attn2 @ V   (K = 256, N = 32)
    if (wave < 2) {
      int nt = wave;
      v8f acc = {};
#pragma unroll
      for (int kk = 0; kk < 8; ++kk) {
        v16bf aa  = frag_A(Sb + (lane & 15) * N_ + kk * 32, lane);
        v16bf bb2 = frag_contig(VT + ((nt << 4) + (lane & 15)) * N_ + kk * 32 + koff);
        acc = wmma_bf16(aa, bb2, acc);
      }
      int d = (nt << 4) + (lane & 15);
      int abase = (lane & 16) ? 8 : 0;
#pragma unroll
      for (int r = 0; r < 8; ++r) {
        int ag = (mt << 4) + abase + r;
        agent_v[((b * H_ + h) * AP_ + ag) * HD_ + d] = acc[r];
      }
    }
    __syncthreads();
  }
}

// ---------------- stage 2: query <- agent attention ------------------------

__global__ __launch_bounds__(256) void stage2_k(
    const float* __restrict__ qf, const float* __restrict__ a1p,
    const float* __restrict__ agent_v, const float* __restrict__ ab,
    float* __restrict__ attn_out) {
  __shared__ u16b  Q[N_ * HD_];      // 16KB
  __shared__ u16b  A1[AP_ * HD_];    //  4KB
  __shared__ u16b  AVT[HD_ * AP_];   //  4KB  [d][a]
  __shared__ float S[64 * AP_];      // 16KB
  __shared__ u16b  Sb[64 * AP_];     //  8KB
  int b = blockIdx.x / H_, h = blockIdx.x % H_;
  int tid = threadIdx.x, wave = tid >> 5, lane = tid & 31;
  int koff = (lane & 16) ? 16 : 0;

  for (int i = tid; i < N_ * HD_; i += 256)
    Q[i] = f2bf(qf[(b * N_ + (i >> 5)) * C_ + h * HD_ + (i & 31)]);
  for (int i = tid; i < AP_ * HD_; i += 256) {
    int a0 = i >> 5, d = i & 31;
    A1[i] = f2bf(a1p[(b * AP_ + a0) * C_ + h * HD_ + d]);
    AVT[d * AP_ + a0] = f2bf(agent_v[((b * H_ + h) * AP_ + a0) * HD_ + d]);
  }
  __syncthreads();

  for (int mb = 0; mb < 4; ++mb) {            // 64 query rows per block-iter
    for (int tt = 0; tt < 2; ++tt) {
      int t = wave * 2 + tt;
      int mtl = t >> 2, nt = t & 3;
      v16bf a   = frag_A(Q + (mb * 64 + (mtl << 4) + (lane & 15)) * HD_, lane);
      v16bf bfr = frag_contig(A1 + ((nt << 4) + (lane & 15)) * HD_ + koff);
      v8f c = {};
      c = wmma_bf16(a, bfr, c);
      int acol = (nt << 4) + (lane & 15);
      int mbase = (mtl << 4) + ((lane & 16) ? 8 : 0);
#pragma unroll
      for (int r = 0; r < 8; ++r) {
        int ml = mbase + r, ng = mb * 64 + ml;
        S[ml * AP_ + acol] = SCALE_ * c[r] + ab[(h * N_ + ng) * AP_ + acol];
      }
    }
    __syncthreads();
    if (tid < 64) {                            // masked softmax over 64 agents
      float mx = -1e30f;
      for (int j = 0; j < AP_; ++j) mx = fmaxf(mx, S[tid * AP_ + j]);
      float sm = 0.0f;
      for (int j = 0; j < AP_; ++j) {
        float e = __expf(S[tid * AP_ + j] - mx);
        S[tid * AP_ + j] = e; sm += e;
      }
      float inv = 1.0f / sm;
      for (int j = 0; j < AP_; ++j) Sb[tid * AP_ + j] = f2bf(S[tid * AP_ + j] * inv);
    }
    __syncthreads();
    {
      int mtl = wave >> 1, nt = wave & 1;      // 8 tiles (64x32) over 8 waves
      v8f acc = {};
#pragma unroll
      for (int kk = 0; kk < 2; ++kk) {
        v16bf aa  = frag_A(Sb + ((mtl << 4) + (lane & 15)) * AP_ + kk * 32, lane);
        v16bf bb2 = frag_contig(AVT + ((nt << 4) + (lane & 15)) * AP_ + kk * 32 + koff);
        acc = wmma_bf16(aa, bb2, acc);
      }
      int d = (nt << 4) + (lane & 15);
      int mbase = (mtl << 4) + ((lane & 16) ? 8 : 0);
#pragma unroll
      for (int r = 0; r < 8; ++r) {
        int ng = mb * 64 + mbase + r;
        attn_out[(b * N_ + ng) * C_ + h * HD_ + d] = acc[r];
      }
    }
    __syncthreads();
  }
}

// ---------------- depthwise 3x3 on q + residual add, emit bf16 -------------

__global__ __launch_bounds__(256) void dwadd_k(
    const float* __restrict__ attn, const float* __restrict__ qf,
    const float* __restrict__ dwc_w, const float* __restrict__ dwc_b,
    u16b* __restrict__ xp) {
  int t = blockIdx.x * 256 + threadIdx.x;
  if (t >= B_ * N_ * C_) return;
  int c = t % C_, n = (t / C_) % N_, b = t / (C_ * N_);
  int hh = n >> 4, ww = n & 15;
  float s = dwc_b[c];
#pragma unroll
  for (int dy = 0; dy < 3; ++dy) {
    int y = hh + dy - 1;
    if (y < 0 || y > 15) continue;
#pragma unroll
    for (int dx = 0; dx < 3; ++dx) {
      int x = ww + dx - 1;
      if (x < 0 || x > 15) continue;
      s += qf[(b * N_ + y * 16 + x) * C_ + c] * dwc_w[c * 9 + dy * 3 + dx];
    }
  }
  xp[t] = f2bf(attn[t] + s);
}

// ---------------- launch ----------------------------------------------------

extern "C" void kernel_launch(void* const* d_in, const int* in_sizes, int n_in,
                              void* d_out, int out_size, void* d_ws, size_t ws_size,
                              hipStream_t stream) {
  const float* x_F1 = (const float*)d_in[0];
  const float* x_C2 = (const float*)d_in[1];
  const float* x_C1 = (const float*)d_in[2];
  const float* Wq = (const float*)d_in[3];  const float* bq = (const float*)d_in[4];
  const float* Wk = (const float*)d_in[5];  const float* bk = (const float*)d_in[6];
  const float* Wv = (const float*)d_in[7];  const float* bv = (const float*)d_in[8];
  const float* Wa1 = (const float*)d_in[9]; const float* ba1 = (const float*)d_in[10];
  const float* Wa2 = (const float*)d_in[11];const float* ba2 = (const float*)d_in[12];
  const float* an_bias = (const float*)d_in[13];
  const float* na_bias = (const float*)d_in[14];
  const float* ah_bias = (const float*)d_in[15];
  const float* aw_bias = (const float*)d_in[16];
  const float* ha_bias = (const float*)d_in[17];
  const float* wa_bias = (const float*)d_in[18];
  const float* dwc_w = (const float*)d_in[19];
  const float* dwc_b = (const float*)d_in[20];
  const float* Wp = (const float*)d_in[21]; const float* bp = (const float*)d_in[22];
  (void)in_sizes; (void)n_in; (void)out_size; (void)ws_size;

  const size_t SZ = (size_t)B_ * N_ * C_;        // 12,582,912 elems
  size_t off = 0;
  auto alloc = [&](size_t bytes) -> void* {
    void* p = (char*)d_ws + off;
    off += (bytes + 255) & ~((size_t)255);
    return p;
  };
  u16b* Xq  = (u16b*)alloc(SZ * 2);
  u16b* Xk  = (u16b*)alloc(SZ * 2);
  u16b* Xv  = (u16b*)alloc(SZ * 2);
  u16b* Xc1 = (u16b*)alloc(SZ * 2);
  u16b* Xc2 = (u16b*)alloc(SZ * 2);
  u16b* Wpk = (u16b*)alloc((size_t)6 * 36864 * 2);  // 6 packed weight matrices
  float* qf    = (float*)alloc(SZ * 4);
  u16b*  kbuf  = (u16b*)alloc(SZ * 2);
  u16b*  vbuf  = (u16b*)alloc(SZ * 2);
  float* a1lin = (float*)alloc(SZ * 4);
  float* a2lin = (float*)alloc(SZ * 4);
  float* a1p   = (float*)alloc((size_t)B_ * AP_ * C_ * 4);
  float* a2p   = (float*)alloc((size_t)B_ * AP_ * C_ * 4);
  float* pbw   = (float*)alloc((size_t)H_ * AG_ * N_ * 4);
  float* abw   = (float*)alloc((size_t)H_ * N_ * AP_ * 4);
  float* agv   = (float*)alloc((size_t)B_ * H_ * AP_ * HD_ * 4);
  float* attn  = a2lin;            // alias: a2lin dead after pooling
  u16b*  xp    = (u16b*)a1lin;     // alias: a1lin dead after pooling

  const int EL_BLK = (int)(SZ / 256);            // 49152

  prep_k<<<EL_BLK, 256, 0, stream>>>(x_F1, x_C2, x_C1, Xq, Xk, Xv, Xc1, Xc2);

  const float* Ws[6] = {Wq, Wk, Wv, Wa1, Wa2, Wp};
  for (int i = 0; i < 6; ++i)
    pack_w_k<<<144, 256, 0, stream>>>(Ws[i], Wpk + (size_t)i * 36864);

  const int GEMM_BLK = (65536 / 16) / 8;         // 512 blocks, 1 strip/wave
  gemm2_k<0><<<GEMM_BLK, 256, 0, stream>>>(Xq,  Wpk + 0 * 36864, bq,  qf,    B_ * N_);
  gemm2_k<1><<<GEMM_BLK, 256, 0, stream>>>(Xk,  Wpk + 1 * 36864, bk,  kbuf,  B_ * N_);
  gemm2_k<1><<<GEMM_BLK, 256, 0, stream>>>(Xv,  Wpk + 2 * 36864, bv,  vbuf,  B_ * N_);
  gemm2_k<0><<<GEMM_BLK, 256, 0, stream>>>(Xc1, Wpk + 3 * 36864, ba1, a1lin, B_ * N_);
  gemm2_k<0><<<GEMM_BLK, 256, 0, stream>>>(Xc2, Wpk + 4 * 36864, ba2, a2lin, B_ * N_);

  pool_k<<<(B_ * AP_ * C_ + 255) / 256, 256, 0, stream>>>(a1lin, a1p);
  pool_k<<<(B_ * AP_ * C_ + 255) / 256, 256, 0, stream>>>(a2lin, a2p);

  bias_pb_k<<<(H_ * AG_ * N_ + 255) / 256, 256, 0, stream>>>(an_bias, ah_bias, aw_bias, pbw);
  bias_ab_k<<<(H_ * N_ * AP_ + 255) / 256, 256, 0, stream>>>(na_bias, ha_bias, wa_bias, abw);

  stage1_k<<<B_ * H_, 256, 0, stream>>>(a2p, kbuf, vbuf, pbw, agv);
  stage2_k<<<B_ * H_, 256, 0, stream>>>(qf, a1p, agv, abw, attn);

  dwadd_k<<<EL_BLK, 256, 0, stream>>>(attn, qf, dwc_w, dwc_b, xp);

  gemm2_k<0><<<GEMM_BLK, 256, 0, stream>>>(xp, Wpk + 5 * 36864, bp, d_out, B_ * N_);
}
